// Attention_ori_83786222010422
// MI455X (gfx1250) — matrix operand order repaired and checked by reading, not compiled, not run on hardware
//
#include <hip/hip_runtime.h>
#include <hip/hip_bf16.h>

// ---------------------------------------------------------------------------
// Channel-wise ("transposed") attention for MI455X (gfx1250, wave32, WMMA).
// All GEMMs via v_wmma_f32_16x16x32_bf16 with f32 accumulation.
// ---------------------------------------------------------------------------

typedef __attribute__((ext_vector_type(16))) __bf16 v16bf;
typedef __attribute__((ext_vector_type(8)))  float  v8f;

#define NTOK  16384   // H*W tokens per batch
#define CDIM  256     // channels
#define BATCH 8
#define HEADS 8
#define DH    32

// Hard scheduling fence: forbid the backend from sinking loads past this
// point (keeps the load clause intact, then back-to-back WMMAs).
#if defined(__has_builtin)
#  if __has_builtin(__builtin_amdgcn_sched_barrier)
#    define SCHED_FENCE() __builtin_amdgcn_sched_barrier(0)
#  endif
#endif
#ifndef SCHED_FENCE
#  define SCHED_FENCE()
#endif

__device__ __forceinline__ float bf2f(unsigned short u) {
    return __uint_as_float(((unsigned)u) << 16);
}
__device__ __forceinline__ unsigned short f2bf(float f) {
    unsigned u = __float_as_uint(f);
    u += 0x7FFFu + ((u >> 16) & 1u);          // round-to-nearest-even
    return (unsigned short)(u >> 16);
}

union Frag {
    v16bf v;
    uint4 q[2];
    unsigned short u[16];
};

// ---------------------------------------------------------------------------
// Kernel 0: convert + transpose weights to bf16, N-major (row = out-channel,
// col = contraction index) so WMMA B-fragments load contiguously.
// ---------------------------------------------------------------------------
__global__ void k_pack_weights(const float* __restrict__ Wq, const float* __restrict__ Wk,
                               const float* __restrict__ Wv, const float* __restrict__ Wp,
                               unsigned short* __restrict__ wqt, unsigned short* __restrict__ wkt,
                               unsigned short* __restrict__ wvt, unsigned short* __restrict__ wpt) {
    int idx = blockIdx.x * blockDim.x + threadIdx.x;   // 4 * 65536 total
    int m   = idx >> 16;
    int e   = idx & 65535;
    int row = e >> 8;     // output row (N index)
    int col = e & 255;    // contraction index
    const float* src = (m == 0) ? Wq : (m == 1) ? Wk : (m == 2) ? Wv : Wp;
    unsigned short* dst = (m == 0) ? wqt : (m == 1) ? wkt : (m == 2) ? wvt : wpt;
    dst[row * 256 + col] = f2bf(src[col * 256 + row]);
}

// ---------------------------------------------------------------------------
// Kernel 1: QKV projection GEMM.  M = B*NTOK = 131072, K = 256, N = 3*256.
// Each wave owns a 16-row M strip; A fragments for the full K=256 live in
// registers (8 x v16bf).  All 8 B fragments of a tile are loaded before a
// sched_barrier, then the 8-WMMA accumulate chain runs back-to-back.
// launch_bounds(128,1) lifts the VGPR cap so A+B fragments (~170 VGPRs)
// stay register-resident (no scratch spills); 2-3 waves/SIMD is enough to
// hide the single clause-wait per tile.
// Q,K stored TRANSPOSED [b,h,d,n] bf16 (packed 16B stores along n),
// V stored natural [b*n, c'] bf16.
// ---------------------------------------------------------------------------
__global__ void __launch_bounds__(128, 1)
k_qkv_gemm(const float* __restrict__ x,
           const unsigned short* __restrict__ wqt,
           const unsigned short* __restrict__ wkt,
           const unsigned short* __restrict__ wvt,
           unsigned short* __restrict__ qt,
           unsigned short* __restrict__ kt,
           unsigned short* __restrict__ vnat) {
    const int lane = threadIdx.x & 31;
    const int wave = threadIdx.x >> 5;
    const long m0  = ((long)blockIdx.x * 4 + wave) * 16;   // global row base
    const int lrow = lane & 15;
    const int hig  = lane >> 4;
    const long b     = m0 / NTOK;
    const long tbase = m0 % NTOK;

    // --- A fragments: 16 rows x 256 K, converted f32 -> bf16 in-register ---
    Frag a[8];
    {
        const float* xrow = x + (m0 + lrow) * CDIM;
        #pragma unroll
        for (int kc = 0; kc < 8; ++kc) {
            const int kbase = kc * 32 + (hig ? 8 : 0);
            #pragma unroll
            for (int g = 0; g < 2; ++g) {
                float4 f0 = *(const float4*)(xrow + kbase + g * 16);
                float4 f1 = *(const float4*)(xrow + kbase + g * 16 + 4);
                a[kc].u[g * 8 + 0] = f2bf(f0.x); a[kc].u[g * 8 + 1] = f2bf(f0.y);
                a[kc].u[g * 8 + 2] = f2bf(f0.z); a[kc].u[g * 8 + 3] = f2bf(f0.w);
                a[kc].u[g * 8 + 4] = f2bf(f1.x); a[kc].u[g * 8 + 5] = f2bf(f1.y);
                a[kc].u[g * 8 + 6] = f2bf(f1.z); a[kc].u[g * 8 + 7] = f2bf(f1.w);
            }
        }
    }

    for (int nt = 0; nt < 48; ++nt) {
        const unsigned short* wt = (nt < 16) ? wqt : (nt < 32) ? wkt : wvt;
        const int n0 = (nt & 15) * 16;
        const unsigned short* wrow = wt + (long)(n0 + lrow) * CDIM + (hig ? 8 : 0);

        // preload all 8 B fragments (one load clause) ...
        Frag bfr[8];
        #pragma unroll
        for (int kc = 0; kc < 8; ++kc) {
            bfr[kc].q[0] = *(const uint4*)(wrow + kc * 32);
            bfr[kc].q[1] = *(const uint4*)(wrow + kc * 32 + 16);
        }
        SCHED_FENCE();   // ... then WMMA chain, no load sinking
        v8f acc = {0.f, 0.f, 0.f, 0.f, 0.f, 0.f, 0.f, 0.f};
        #pragma unroll
        for (int kc = 0; kc < 8; ++kc) {
            acc = __builtin_amdgcn_wmma_f32_16x16x32_bf16(
                      false, a[kc].v, false, bfr[kc].v, (short)0, acc, false, false);
        }
        SCHED_FENCE();

        const int cp = n0 + lrow;                     // column inside 256-block
        if (nt < 32) {
            // transposed store: 8 consecutive n-positions -> one 16B store
            unsigned short* dstb = (nt < 16) ? qt : kt;
            const int h = cp >> 5, i = cp & 31;
            const long off = (((b * HEADS + h) * DH + i) * (long)NTOK) + tbase + hig * 8;
            unsigned pr[4];
            #pragma unroll
            for (int r = 0; r < 4; ++r)
                pr[r] = (unsigned)f2bf(acc[2 * r]) | ((unsigned)f2bf(acc[2 * r + 1]) << 16);
            uint4 pk; pk.x = pr[0]; pk.y = pr[1]; pk.z = pr[2]; pk.w = pr[3];
            *(uint4*)(dstb + off) = pk;
        } else {
            #pragma unroll
            for (int r = 0; r < 8; ++r) {
                const long row = m0 + hig * 8 + r;
                vnat[row * CDIM + cp] = f2bf(acc[r]);
            }
        }
    }
}

// ---------------------------------------------------------------------------
// Kernel 2: per (b,h) Gram S = K Q^T over n, fold the l2-norms into the
// logits, softmax over j, store attn^T bf16.
// 256 threads = 8 waves: 4 output tiles x 2 n-segments; each wave streams
// 8192 n with a x4-unrolled, preloaded WMMA loop; partials summed in LDS.
// ---------------------------------------------------------------------------
__global__ void k_gram_softmax(const unsigned short* __restrict__ qt,
                               const unsigned short* __restrict__ kt,
                               const float* __restrict__ rescale,
                               unsigned short* __restrict__ attnT) {
    __shared__ float s_norm[64];        // [0,32): ||k_i||^2, [32,64): ||q_j||^2
    __shared__ float s_S[2][32 * 32];   // one partial Gram per n-segment
    const int bh   = blockIdx.x;        // b*8 + h
    const int h    = bh & 7;
    const int tid  = threadIdx.x;       // 256 threads = 8 waves
    const int lane = tid & 31;
    const int wave = tid >> 5;

    if (tid < 64) s_norm[tid] = 0.f;
    __syncthreads();

    // --- phase 1: row sum-of-squares (4 threads per row) ---
    {
        const int row = tid >> 2;       // 0..63
        const int qtr = tid & 3;
        const unsigned short* base = (row < 32)
            ? (kt + ((long)bh * 32 + row) * NTOK)
            : (qt + ((long)bh * 32 + (row - 32)) * NTOK);
        const uint4* p = (const uint4*)(base + qtr * 4096);
        float s = 0.f;
        for (int it = 0; it < 512; ++it) {
            uint4 d = p[it];
            unsigned w[4] = {d.x, d.y, d.z, d.w};
            #pragma unroll
            for (int e = 0; e < 4; ++e) {
                float f0 = bf2f((unsigned short)(w[e] & 0xFFFFu));
                float f1 = bf2f((unsigned short)(w[e] >> 16));
                s += f0 * f0 + f1 * f1;
            }
        }
        atomicAdd(&s_norm[row], s);
    }
    __syncthreads();

    // --- phase 2: wave-private 16x16 tile over one n-segment ---
    const int tile = wave & 3;          // which 16x16 tile of S
    const int seg  = wave >> 2;         // which half of the n range
    const int i0 = (tile >> 1) * 16, j0 = (tile & 1) * 16;
    const int lrow = lane & 15, hig = lane >> 4;
    v8f acc = {0.f, 0.f, 0.f, 0.f, 0.f, 0.f, 0.f, 0.f};
    const unsigned short* arow = kt + ((long)bh * 32 + i0 + lrow) * NTOK
                               + seg * 8192 + (hig ? 8 : 0);
    const unsigned short* brow = qt + ((long)bh * 32 + j0 + lrow) * NTOK
                               + seg * 8192 + (hig ? 8 : 0);
    for (int n0 = 0; n0 < 8192; n0 += 128) {
        __builtin_prefetch(arow + n0 + 128, 0, 1);
        __builtin_prefetch(brow + n0 + 128, 0, 1);
        Frag af[4], bfr[4];
        #pragma unroll
        for (int u = 0; u < 4; ++u) {
            af[u].q[0]  = *(const uint4*)(arow + n0 + u * 32);
            af[u].q[1]  = *(const uint4*)(arow + n0 + u * 32 + 16);
            bfr[u].q[0] = *(const uint4*)(brow + n0 + u * 32);
            bfr[u].q[1] = *(const uint4*)(brow + n0 + u * 32 + 16);
        }
        SCHED_FENCE();
        #pragma unroll
        for (int u = 0; u < 4; ++u) {
            acc = __builtin_amdgcn_wmma_f32_16x16x32_bf16(
                      false, af[u].v, false, bfr[u].v, (short)0, acc, false, false);
        }
        SCHED_FENCE();
    }
    #pragma unroll
    for (int r = 0; r < 8; ++r)
        s_S[seg][(i0 + hig * 8 + r) * 32 + (j0 + lrow)] = acc[r];
    __syncthreads();

    // --- phase 3: scale by 1/(||k_i|| ||q_j||) * rescale, softmax over j ---
    if (tid < 32) {
        const int i = tid;
        const float rs = rescale[h];
        const float nk = fmaxf(sqrtf(s_norm[i]), 1e-12f);
        float rowv[32];
        float mx = -3.4e38f;
        for (int j = 0; j < 32; ++j) {
            const float nq = fmaxf(sqrtf(s_norm[32 + j]), 1e-12f);
            const float v = (s_S[0][i * 32 + j] + s_S[1][i * 32 + j]) / (nk * nq) * rs;
            rowv[j] = v;
            mx = fmaxf(mx, v);
        }
        float sum = 0.f;
        for (int j = 0; j < 32; ++j) { rowv[j] = __expf(rowv[j] - mx); sum += rowv[j]; }
        const float inv = 1.f / sum;
        for (int j = 0; j < 32; ++j)
            attnT[(long)bh * 1024 + j * 32 + i] = f2bf(rowv[j] * inv);  // transposed
    }
}

// ---------------------------------------------------------------------------
// Kernel 3a: Weff_b[h*32+j, c] = sum_i attn[b,h,i,j] * Wp[h*32+i, c]
// (folds the output projection through the attention). One WMMA per tile.
// Stored transposed bf16: weffT[b][c][cc].
// ---------------------------------------------------------------------------
__global__ void k_weff(const unsigned short* __restrict__ attnT,
                       const unsigned short* __restrict__ wpt,
                       unsigned short* __restrict__ weffT) {
    const int bh = blockIdx.x;          // b*8 + h
    const int b  = bh >> 3, h = bh & 7;
    const int lane = threadIdx.x & 31, wave = threadIdx.x >> 5;
    const int lrow = lane & 15, hig = lane >> 4;
    const int j0 = (wave & 1) * 16;

    Frag af;
    {
        const unsigned short* ar = attnT + (long)bh * 1024 + (j0 + lrow) * 32 + (hig ? 8 : 0);
        af.q[0] = *(const uint4*)(ar);
        af.q[1] = *(const uint4*)(ar + 16);
    }
    for (int ct = 0; ct < 8; ++ct) {
        const int c0 = ((wave >> 1) + ct * 2) * 16;
        const unsigned short* br = wpt + (long)(c0 + lrow) * CDIM + h * 32 + (hig ? 8 : 0);
        Frag bfr;
        bfr.q[0] = *(const uint4*)(br);
        bfr.q[1] = *(const uint4*)(br + 16);
        v8f acc = {0.f, 0.f, 0.f, 0.f, 0.f, 0.f, 0.f, 0.f};
        acc = __builtin_amdgcn_wmma_f32_16x16x32_bf16(
                  false, af.v, false, bfr.v, (short)0, acc, false, false);
        unsigned pr[4];
        #pragma unroll
        for (int r = 0; r < 4; ++r)
            pr[r] = (unsigned)f2bf(acc[2 * r]) | ((unsigned)f2bf(acc[2 * r + 1]) << 16);
        uint4 pk; pk.x = pr[0]; pk.y = pr[1]; pk.z = pr[2]; pk.w = pr[3];
        unsigned short* dst = weffT + (long)b * 65536 + (long)(c0 + lrow) * CDIM
                            + h * 32 + j0 + hig * 8;
        *(uint4*)dst = pk;
    }
}

// ---------------------------------------------------------------------------
// Kernel 3b: final GEMM  out[row, c] = sum_cc vnat[row, cc] * Weff_b[cc, c] + bp[c]
// Same structure as kernel 1 (preloaded B fragments + sched fence +
// launch_bounds to avoid spills); f32 out.
// ---------------------------------------------------------------------------
__global__ void __launch_bounds__(128, 1)
k_out_gemm(const unsigned short* __restrict__ vnat,
           const unsigned short* __restrict__ weffT,
           const float* __restrict__ bp,
           float* __restrict__ out) {
    const int lane = threadIdx.x & 31, wave = threadIdx.x >> 5;
    const long m0  = ((long)blockIdx.x * 4 + wave) * 16;
    const int lrow = lane & 15, hig = lane >> 4;
    const long b = m0 / NTOK;

    Frag a[8];
    {
        const unsigned short* xr = vnat + (m0 + lrow) * CDIM + (hig ? 8 : 0);
        #pragma unroll
        for (int kc = 0; kc < 8; ++kc) {
            a[kc].q[0] = *(const uint4*)(xr + kc * 32);
            a[kc].q[1] = *(const uint4*)(xr + kc * 32 + 16);
        }
    }
    const unsigned short* wb = weffT + b * 65536;
    for (int nt = 0; nt < 16; ++nt) {
        const int c0 = nt * 16;
        const unsigned short* wr = wb + (long)(c0 + lrow) * CDIM + (hig ? 8 : 0);

        Frag bfr[8];
        #pragma unroll
        for (int kc = 0; kc < 8; ++kc) {
            bfr[kc].q[0] = *(const uint4*)(wr + kc * 32);
            bfr[kc].q[1] = *(const uint4*)(wr + kc * 32 + 16);
        }
        SCHED_FENCE();
        v8f acc = {0.f, 0.f, 0.f, 0.f, 0.f, 0.f, 0.f, 0.f};
        #pragma unroll
        for (int kc = 0; kc < 8; ++kc) {
            acc = __builtin_amdgcn_wmma_f32_16x16x32_bf16(
                      false, a[kc].v, false, bfr[kc].v, (short)0, acc, false, false);
        }
        SCHED_FENCE();
        const float bias = bp[c0 + lrow];
        #pragma unroll
        for (int r = 0; r < 8; ++r) {
            const long row = m0 + hig * 8 + r;
            out[row * CDIM + (c0 + lrow)] = acc[r] + bias;
        }
    }
}

// ---------------------------------------------------------------------------
extern "C" void kernel_launch(void* const* d_in, const int* in_sizes, int n_in,
                              void* d_out, int out_size, void* d_ws, size_t ws_size,
                              hipStream_t stream) {
    (void)in_sizes; (void)n_in; (void)out_size; (void)ws_size;

    const float* x       = (const float*)d_in[0];
    const float* Wq      = (const float*)d_in[1];
    const float* Wk      = (const float*)d_in[2];
    const float* Wv      = (const float*)d_in[3];
    const float* rescale = (const float*)d_in[4];
    const float* Wp      = (const float*)d_in[5];
    const float* bp      = (const float*)d_in[6];
    float* out = (float*)d_out;

    // workspace carve-up (element counts are ushort/bf16 unless noted)
    unsigned short* ws = (unsigned short*)d_ws;
    size_t o = 0;
    unsigned short* wqt   = ws + o; o += 256 * 256;                 // Wq^T bf16
    unsigned short* wkt   = ws + o; o += 256 * 256;                 // Wk^T bf16
    unsigned short* wvt   = ws + o; o += 256 * 256;                 // Wv^T bf16
    unsigned short* wpt   = ws + o; o += 256 * 256;                 // Wp^T bf16
    unsigned short* qt    = ws + o; o += (size_t)BATCH * HEADS * DH * NTOK; // Q [b,h,d,n]
    unsigned short* kt    = ws + o; o += (size_t)BATCH * HEADS * DH * NTOK; // K [b,h,d,n]
    unsigned short* vnat  = ws + o; o += (size_t)BATCH * NTOK * CDIM;       // V natural
    unsigned short* attnT = ws + o; o += (size_t)64 * 32 * 32;      // attn^T bf16
    unsigned short* weffT = ws + o; o += (size_t)BATCH * 256 * 256; // Weff^T bf16

    k_pack_weights<<<1024, 256, 0, stream>>>(Wq, Wk, Wv, Wp, wqt, wkt, wvt, wpt);
    k_qkv_gemm    <<<2048, 128, 0, stream>>>(x, wqt, wkt, wvt, qt, kt, vnat);
    k_gram_softmax<<<  64, 256, 0, stream>>>(qt, kt, rescale, attnT);
    k_weff        <<<  64, 128, 0, stream>>>(attnT, wpt, weffT);
    k_out_gemm    <<<2048, 128, 0, stream>>>(vnat, weffT, bp, out);
}
